// Sequence_3058016714762
// MI455X (gfx1250) — matrix operand order repaired and checked by reading, not compiled, not run on hardware
//
#include <hip/hip_runtime.h>

// Problem dims (compile-time)
#define Bdim   512
#define Tdim   2048
#define INdim  16
#define Hdim   128
#define OUTdim 8

// LDS strides (elements), padded to avoid bank conflicts
#define XAS 40        // x A-buffer, bf16, K padded 16->32 (+bank pad)
#define HAS 136       // h A-buffers, bf16, K=128 (+8 pad)
#define HBUF (16*HAS) // elements per h double-buffer slice (2176)

typedef __attribute__((ext_vector_type(16))) __bf16 v16bf;
typedef __attribute__((ext_vector_type(8)))  __bf16 v8bf;
typedef __attribute__((ext_vector_type(8)))  float  v8f;

// Hardware approximate reciprocal (v_rcp_f32) — avoids IEEE div expansion.
__device__ __forceinline__ float fast_rcp(float x) {
  return __builtin_amdgcn_rcpf(x);
}
__device__ __forceinline__ float sigmoidf_(float x) {
  float e = __expf(-x);
  return fast_rcp(1.0f + e);
}
__device__ __forceinline__ float tanhf_(float x) {
  float ax = fabsf(x);
  float e  = __expf(-2.0f * ax);
  float r  = (1.0f - e) * fast_rcp(1.0f + e);
  return copysignf(r, x);
}

__device__ __forceinline__ v8f wmma_bf16(v16bf a, v16bf b, v8f c) {
  // D = A(16x32 bf16) * B(32x16 bf16) + C(16x16 f32)
  return __builtin_amdgcn_wmma_f32_16x16x32_bf16(
      /*neg_a=*/false, a, /*neg_b=*/false, b,
      /*c_mod=*/(short)0, c, /*reuse_a=*/false, /*reuse_b=*/false);
}

// Build a B-matrix fragment (32x16 slice of W^T) from global f32 weights -> bf16.
__device__ __forceinline__ v16bf load_b_frag(const float* __restrict__ W, int rowlen,
                                             int g, int k0, int koff, int realK) {
  v16bf b;
#pragma unroll
  for (int i = 0; i < 16; i++) {
    int k = k0 + koff + i;
    float v = (k < realK) ? W[g * rowlen + k] : 0.0f;
    b[i] = (__bf16)v;
  }
  return b;
}

// Load a 16x32 bf16 A fragment from an LDS row-major buffer (two aligned b128s).
__device__ __forceinline__ v16bf load_a_frag(const __bf16* base, int stride,
                                             int row, int k0, int l8) {
  const __bf16* p = base + row * stride + k0 + l8;
  v8bf lo = *(const v8bf*)(p);
  v8bf hi = *(const v8bf*)(p + 8);
  return __builtin_shufflevector(lo, hi, 0,1,2,3,4,5,6,7,8,9,10,11,12,13,14,15);
}

__global__ void __launch_bounds__(256, 1) gru2_persistent(
    const float* __restrict__ x,
    const float* __restrict__ Wih0, const float* __restrict__ Whh0,
    const float* __restrict__ bih0, const float* __restrict__ bhh0,
    const float* __restrict__ Wih1, const float* __restrict__ Whh1,
    const float* __restrict__ bih1, const float* __restrict__ bhh1,
    const float* __restrict__ Wout, const float* __restrict__ bo_g,
    float* __restrict__ out)
{
  extern __shared__ char smem[];
  __bf16* xA      = (__bf16*)(smem);           // 16*40*2 = 1280
  __bf16* hA0base = (__bf16*)(smem + 1280);    // 2 buffers x 4352 B
  __bf16* hA1base = (__bf16*)(smem + 9984);    // 2 buffers x 4352 B (end 18688)

  const int tid  = threadIdx.x;
  const int lane = tid & 31;
  const int wave = tid >> 5;            // 8 waves; wave w owns columns [w*16, w*16+16)
  const int l15  = lane & 15;
  const int l8   = (lane < 16) ? 0 : 16;  // A-frag K sub-offset
  const int koff = (lane < 16) ? 0 : 16;  // B-frag K sub-offset
  const int rh   = (lane < 16) ? 0 : 8;   // C/D row half
  const int b0   = blockIdx.x * 16;       // batch tile origin
  const int jcol = wave * 16 + l15;       // this lane's hidden column (0..127)

  // ---------------- init LDS ----------------
  for (int i = tid; i < 16 * XAS; i += 256) xA[i] = (__bf16)0.0f;
  for (int i = tid; i < 2 * HBUF; i += 256) {
    hA0base[i] = (__bf16)0.0f;
    hA1base[i] = (__bf16)0.0f;
  }

  // ------- pin all WMMA B-fragments (weights, bf16) in VGPRs -------
  v16bf fih0[3];                               // K=16 padded to 32
  v16bf fhh0[3][4], fih1[3][4], fhh1[3][4];    // K=128, 4 k-steps each
#pragma unroll
  for (int nt = 0; nt < 3; nt++) {
    int g = nt * 128 + jcol;                   // weight row = B column
    fih0[nt] = load_b_frag(Wih0, INdim, g, 0, koff, INdim);
#pragma unroll
    for (int ks = 0; ks < 4; ks++) {
      fhh0[nt][ks] = load_b_frag(Whh0, Hdim, g, ks * 32, koff, Hdim);
      fih1[nt][ks] = load_b_frag(Wih1, Hdim, g, ks * 32, koff, Hdim);
      fhh1[nt][ks] = load_b_frag(Whh1, Hdim, g, ks * 32, koff, Hdim);
    }
  }
  // Output-projection B fragments (Wout^T, N padded 8->16 with zeros); wave 0 uses them.
  v16bf fout[4];
#pragma unroll
  for (int ks = 0; ks < 4; ks++) {
#pragma unroll
    for (int i = 0; i < 16; i++) {
      int k = ks * 32 + koff + i;
      float v = (l15 < OUTdim) ? Wout[l15 * Hdim + k] : 0.0f;
      fout[ks][i] = (__bf16)v;
    }
  }
  const float bo_r = bo_g[l15 & 7];

  // ------- per-lane gate biases (loop-invariant) in registers -------
  const float brj0 = bih0[jcol] + bhh0[jcol];
  const float bzj0 = bih0[128 + jcol] + bhh0[128 + jcol];
  const float bxj0 = bih0[256 + jcol];
  const float bhj0 = bhh0[256 + jcol];
  const float brj1 = bih1[jcol] + bhh1[jcol];
  const float bzj1 = bih1[128 + jcol] + bhh1[128 + jcol];
  const float bxj1 = bih1[256 + jcol];
  const float bhj1 = bhh1[256 + jcol];

  // Hidden state in registers, C-matrix layout: hst[v] = h[rh+v][jcol], f32.
  v8f hst0 = {}, hst1 = {};

  // Prefetch x(t=0) into a register.
  const int xr = tid >> 4, xk = tid & 15;
  float xreg = x[((size_t)(b0 + xr) * Tdim + 0) * INdim + xk];

  __syncthreads();

#pragma unroll 1
  for (int t = 0; t < Tdim; ++t) {
    const int p = t & 1;
    __bf16* h0r = hA0base + p * HBUF;        // old h0 (A-side)
    __bf16* h0w = hA0base + (p ^ 1) * HBUF;  // new h0
    __bf16* h1r = hA1base + p * HBUF;        // old h1 (= h1 of step t-1)
    __bf16* h1w = hA1base + (p ^ 1) * HBUF;  // new h1

    // ---- window A: stage x(t); wave0 emits out(t-1) via WMMA ----
    xA[xr * XAS + xk] = (__bf16)xreg;
    if (wave == 0 && t > 0) {
      v8f acc = {};
#pragma unroll
      for (int ks = 0; ks < 4; ks++)
        acc = wmma_bf16(load_a_frag(h1r, HAS, l15, ks * 32, l8), fout[ks], acc);
      if (l15 < OUTdim) {
#pragma unroll
        for (int v = 0; v < 8; v++)
          out[((size_t)(b0 + rh + v) * Tdim + (t - 1)) * OUTdim + l15] = acc[v] + bo_r;
      }
    }
    __syncthreads();

    // ---- phase B: layer-0 GEMM + in-register gates; issue x(t+1) prefetch ----
    {
      int tn = (t + 1 < Tdim) ? t + 1 : t;
      xreg = x[((size_t)(b0 + xr) * Tdim + tn) * INdim + xk];
    }
    {
      v8f aR = {}, aZ = {}, aNX = {}, aNH = {};
      v16bf ax = load_a_frag(xA, XAS, l15, 0, l8);
      aR  = wmma_bf16(ax, fih0[0], aR);
      aZ  = wmma_bf16(ax, fih0[1], aZ);
      aNX = wmma_bf16(ax, fih0[2], aNX);
#pragma unroll
      for (int ks = 0; ks < 4; ks++) {
        v16bf ah = load_a_frag(h0r, HAS, l15, ks * 32, l8);
        aR  = wmma_bf16(ah, fhh0[0][ks], aR);
        aZ  = wmma_bf16(ah, fhh0[1][ks], aZ);
        aNH = wmma_bf16(ah, fhh0[2][ks], aNH);
      }
      // gates fused in C-layout registers; write only bf16 copy for next GEMM
#pragma unroll
      for (int v = 0; v < 8; v++) {
        float r = sigmoidf_(aR[v] + brj0);
        float z = sigmoidf_(aZ[v] + bzj0);
        float n = tanhf_(aNX[v] + bxj0 + r * (aNH[v] + bhj0));
        float h = fmaf(z, hst0[v] - n, n);   // (1-z)*n + z*h
        hst0[v] = h;
        h0w[(rh + v) * HAS + jcol] = (__bf16)h;
      }
    }
    __syncthreads();

    // ---- phase C: layer-1 GEMM (xg1 from new h0, hg1 from old h1) + gates ----
    {
      v8f aR = {}, aZ = {}, aNX = {}, aNH = {};
#pragma unroll
      for (int ks = 0; ks < 4; ks++) {
        v16bf a0 = load_a_frag(h0w, HAS, l15, ks * 32, l8);
        aR  = wmma_bf16(a0, fih1[0][ks], aR);
        aZ  = wmma_bf16(a0, fih1[1][ks], aZ);
        aNX = wmma_bf16(a0, fih1[2][ks], aNX);
      }
#pragma unroll
      for (int ks = 0; ks < 4; ks++) {
        v16bf a1 = load_a_frag(h1r, HAS, l15, ks * 32, l8);
        aR  = wmma_bf16(a1, fhh1[0][ks], aR);
        aZ  = wmma_bf16(a1, fhh1[1][ks], aZ);
        aNH = wmma_bf16(a1, fhh1[2][ks], aNH);
      }
#pragma unroll
      for (int v = 0; v < 8; v++) {
        float r = sigmoidf_(aR[v] + brj1);
        float z = sigmoidf_(aZ[v] + bzj1);
        float n = tanhf_(aNX[v] + bxj1 + r * (aNH[v] + bhj1));
        float h = fmaf(z, hst1[v] - n, n);
        hst1[v] = h;
        h1w[(rh + v) * HAS + jcol] = (__bf16)h;
      }
    }
    __syncthreads();
  }

  // ---- epilogue: out(T-1); last written h1 buffer is parity (Tdim & 1) ----
  if (wave == 0) {
    __bf16* h1last = hA1base + (Tdim & 1) * HBUF;
    v8f acc = {};
#pragma unroll
    for (int ks = 0; ks < 4; ks++)
      acc = wmma_bf16(load_a_frag(h1last, HAS, l15, ks * 32, l8), fout[ks], acc);
    if (l15 < OUTdim) {
#pragma unroll
      for (int v = 0; v < 8; v++)
        out[((size_t)(b0 + rh + v) * Tdim + (Tdim - 1)) * OUTdim + l15] = acc[v] + bo_r;
    }
  }
}

extern "C" void kernel_launch(void* const* d_in, const int* in_sizes, int n_in,
                              void* d_out, int out_size, void* d_ws, size_t ws_size,
                              hipStream_t stream) {
  (void)in_sizes; (void)n_in; (void)d_ws; (void)ws_size; (void)out_size;
  const float* x    = (const float*)d_in[0];
  const float* Wih0 = (const float*)d_in[1];
  const float* Whh0 = (const float*)d_in[2];
  const float* bih0 = (const float*)d_in[3];
  const float* bhh0 = (const float*)d_in[4];
  const float* Wih1 = (const float*)d_in[5];
  const float* Whh1 = (const float*)d_in[6];
  const float* bih1 = (const float*)d_in[7];
  const float* bhh1 = (const float*)d_in[8];
  const float* Wout = (const float*)d_in[9];
  const float* bout = (const float*)d_in[10];
  float* out = (float*)d_out;

  constexpr size_t smem_bytes = 18688;  // ~18 KB dynamic LDS
  gru2_persistent<<<Bdim / 16, 256, smem_bytes, stream>>>(
      x, Wih0, Whh0, bih0, bhh0, Wih1, Whh1, bih1, bhh1, Wout, bout, out);
}
